// GCN_90177133347179
// MI455X (gfx1250) — matrix operand order repaired and checked by reading, not compiled, not run on hardware
//
#include <hip/hip_runtime.h>
#include <hip/hip_fp16.h>

typedef __attribute__((ext_vector_type(16))) _Float16 v16h;
typedef __attribute__((ext_vector_type(8)))  _Float16 v8h;
typedef __attribute__((ext_vector_type(8)))  float    v8f;

// ---------------------------------------------------------------- degree/norm
__global__ __launch_bounds__(256) void k_init_deg(float* deg, int N) {
    int i = blockIdx.x * 256 + threadIdx.x;
    if (i < N) deg[i] = 1.0f;              // self-loop contributes 1
}

__global__ __launch_bounds__(256) void k_deg_edges(const int* __restrict__ dst,
                                                   float* __restrict__ deg, int E) {
    int e = blockIdx.x * 256 + threadIdx.x;
    if (e < E) atomicAdd(&deg[dst[e]], 1.0f);
}

__global__ __launch_bounds__(256) void k_dinv(const float* __restrict__ deg,
                                              float* __restrict__ dinv, int N) {
    int i = blockIdx.x * 256 + threadIdx.x;
    if (i < N) {
        float d = deg[i];
        dinv[i] = d > 0.0f ? rsqrtf(d) : 0.0f;
    }
}

// ---------------------------------------------------------------- conversions
__global__ __launch_bounds__(256) void k_f32_to_f16(const float* __restrict__ in,
                                                    _Float16* __restrict__ out,
                                                    long n) {
    long i = (long)blockIdx.x * 256 + threadIdx.x;
    long stride = (long)gridDim.x * 256;
    for (; i < n; i += stride) out[i] = (_Float16)in[i];
}

// W [K,Nc] f32 row-major  ->  Wt [Nc,K] f16 row-major (transposed)
__global__ __launch_bounds__(256) void k_wT_f16(const float* __restrict__ W,
                                                _Float16* __restrict__ Wt,
                                                int K, int Nc) {
    int idx = blockIdx.x * 256 + threadIdx.x;
    if (idx < K * Nc) {
        int k = idx / Nc, n = idx % Nc;
        Wt[(size_t)n * K + k] = (_Float16)W[idx];
    }
}

// act16 = f16(a + b)   (residual h3 + h1)
__global__ __launch_bounds__(256) void k_add_f16(const float* __restrict__ a,
                                                 const float* __restrict__ b,
                                                 _Float16* __restrict__ out, long n) {
    long i = (long)blockIdx.x * 256 + threadIdx.x;
    long stride = (long)gridDim.x * 256;
    for (; i < n; i += stride) out[i] = (_Float16)(a[i] + b[i]);
}

// ---------------------------------------------------------------- WMMA GEMM
// C[M,Nc] f32 = A[M,K] f16 row-major  @  Bt[Nc,K] f16 (i.e. B[K,Nc] transposed)
// Block: 256 threads = 8 waves; wave w owns columns n0 = bx*128 + w*16 and
// sweeps TM row tiles (reusing its B fragment).  K % 64 == 0, Nc % 128 == 0.
// TM is a compile-time constant so the inner loop is branch-free straight-line
// code: no accumulator phi-copies around the WMMAs.
__device__ __forceinline__ v16h load_afrag(const _Float16* __restrict__ A,
                                           size_t rowK, int k0, int half) {
    // A fragment: lane holds row m0+r; half 0 -> K {0..15}, half 1 -> K {16..31}
    const _Float16* ap = A + rowK + k0 + half * 16;
    const v8h alo = *(const v8h*)(ap);
    const v8h ahi = *(const v8h*)(ap + 8);
    v16h a;
#pragma unroll
    for (int h = 0; h < 8; ++h) { a[h] = alo[h]; a[h + 8] = ahi[h]; }
    return a;
}

template <int TM>
__device__ __forceinline__ void gemm_tiles(const _Float16* __restrict__ A,
                                           const _Float16* __restrict__ Bt,
                                           float* __restrict__ C,
                                           int K, int Nc, int mbase, int n0,
                                           int half, int r) {
    v8f acc[TM] = {};
    const size_t bRow = (size_t)(n0 + r) * K + half * 16;
    size_t aRow[TM];
#pragma unroll
    for (int t = 0; t < TM; ++t) aRow[t] = (size_t)(mbase + t * 16 + r) * K;

#pragma unroll 2
    for (int k0 = 0; k0 < K; k0 += 32) {
        // B fragment: lane holds column n0+r, halves h -> K = half*16 + h
        const v16h bfrag = *(const v16h*)(Bt + bRow + k0);
#pragma unroll
        for (int t = 0; t < TM; ++t) {
            const v16h afrag = load_afrag(A, aRow[t], k0, half);
            acc[t] = __builtin_amdgcn_wmma_f32_16x16x32_f16(
                false, afrag, false, bfrag, (short)0, acc[t], false, false);
        }
    }

#pragma unroll
    for (int t = 0; t < TM; ++t) {
        const int row = mbase + t * 16 + half * 8;  // C: vgpr v -> M=half*8+v, N=n0+r
#pragma unroll
        for (int v = 0; v < 8; ++v)
            C[(size_t)(row + v) * Nc + (n0 + r)] = acc[t][v];
    }
}

__global__ __launch_bounds__(256) void k_gemm_wmma(const _Float16* __restrict__ A,
                                                   const _Float16* __restrict__ Bt,
                                                   float* __restrict__ C,
                                                   int M, int K, int Nc) {
    const int lane = threadIdx.x & 31;
    const int wave = threadIdx.x >> 5;      // 0..7
    const int half = lane >> 4;             // 0/1
    const int r    = lane & 15;

    const int n0    = blockIdx.x * 128 + wave * 16;
    const int mbase = blockIdx.y * 64;
    const int rem   = M - mbase;            // uniform per block; M % 16 == 0

    if (rem >= 64)      gemm_tiles<4>(A, Bt, C, K, Nc, mbase, n0, half, r);
    else if (rem >= 48) gemm_tiles<3>(A, Bt, C, K, Nc, mbase, n0, half, r);
    else if (rem >= 32) gemm_tiles<2>(A, Bt, C, K, Nc, mbase, n0, half, r);
    else                gemm_tiles<1>(A, Bt, C, K, Nc, mbase, n0, half, r);
}

// ---------------------------------------------------------------- aggregation
// self-loop term initializes agg (also clears stale state each call)
__global__ __launch_bounds__(256) void k_agg_self(const float* __restrict__ dinv,
                                                  const float* __restrict__ xw,
                                                  float* __restrict__ agg, int F) {
    const int i = blockIdx.x;
    const float w = dinv[i] * dinv[i];
    const float* xr = xw + (size_t)i * F;
    float* ar = agg + (size_t)i * F;
    for (int f = threadIdx.x; f < F; f += 256) ar[f] = w * xr[f];
}

// one wave per edge: gather row xw[src], scale, atomic scatter-add into agg[dst]
// xw rows (<=102 MB) are L2-resident (192 MB), so this is L2 traffic, not HBM.
__global__ __launch_bounds__(256) void k_agg_edges(const int* __restrict__ src,
                                                   const int* __restrict__ dst,
                                                   const float* __restrict__ dinv,
                                                   const float* __restrict__ xw,
                                                   float* __restrict__ agg,
                                                   int E, int F) {
    const int e = blockIdx.x * 8 + (threadIdx.x >> 5);
    if (e >= E) return;
    const int lane = threadIdx.x & 31;
    const int s = src[e], d = dst[e];
    const float w = dinv[s] * dinv[d];
    const float* xr = xw + (size_t)s * F;
    float* ar = agg + (size_t)d * F;
    for (int f = lane; f < F; f += 32) atomicAdd(&ar[f], w * xr[f]);
}

// ---------------------------------------------------------------- LN + ELU
__device__ __forceinline__ float block_reduce_sum(float v, float* sm) {
    const int tid = threadIdx.x;
    sm[tid] = v;
    __syncthreads();
    for (int s = 128; s > 0; s >>= 1) {
        if (tid < s) sm[tid] += sm[tid + s];
        __syncthreads();
    }
    const float r = sm[0];
    __syncthreads();
    return r;
}

// row-parallel: t = agg + bias ; LN(t)*g + be ; elu ; optional f32/f16 sinks
__global__ __launch_bounds__(256) void k_ln_elu(const float* __restrict__ agg,
                                                const float* __restrict__ bias,
                                                const float* __restrict__ g,
                                                const float* __restrict__ be,
                                                float* conv_out,     // nullable
                                                float* bn_out,       // nullable
                                                float* f32_out,      // nullable
                                                _Float16* f16_out,   // nullable
                                                int F) {
    __shared__ float sm[256];
    const int row = blockIdx.x;
    const float* x = agg + (size_t)row * F;

    float s = 0.0f;
    for (int f = threadIdx.x; f < F; f += 256) s += x[f] + bias[f];
    const float mu = block_reduce_sum(s, sm) / (float)F;

    float v = 0.0f;
    for (int f = threadIdx.x; f < F; f += 256) {
        const float t = x[f] + bias[f] - mu;
        v += t * t;
    }
    const float var  = block_reduce_sum(v, sm) / (float)F;
    const float rstd = rsqrtf(var + 1e-5f);

    for (int f = threadIdx.x; f < F; f += 256) {
        const float t = x[f] + bias[f];
        if (conv_out) conv_out[(size_t)row * F + f] = t;
        const float y = (t - mu) * rstd * g[f] + be[f];
        if (bn_out) bn_out[(size_t)row * F + f] = y;
        const float z = y > 0.0f ? y : (expf(y) - 1.0f);
        if (f32_out) f32_out[(size_t)row * F + f] = z;
        if (f16_out) f16_out[(size_t)row * F + f] = (_Float16)z;
    }
}

// ---------------------------------------------------------------- classifier
// logits[i,c] = elu(bn[i,:]) @ Wc[:,c] + bc[c]   (F=128, nc=4)
__global__ __launch_bounds__(256) void k_classifier(const float* __restrict__ bn,
                                                    const float* __restrict__ Wc,
                                                    const float* __restrict__ bc,
                                                    float* __restrict__ out,
                                                    int N, int F, int nc) {
    const int i = blockIdx.x * 256 + threadIdx.x;
    if (i >= N) return;
    float acc[4];
    for (int c = 0; c < nc; ++c) acc[c] = bc[c];
    const float* r = bn + (size_t)i * F;
    for (int k = 0; k < F; ++k) {
        float h = r[k];
        h = h > 0.0f ? h : (expf(h) - 1.0f);
        for (int c = 0; c < nc; ++c) acc[c] += h * Wc[k * nc + c];
    }
    for (int c = 0; c < nc; ++c) out[(size_t)i * nc + c] = acc[c];
}

// ---------------------------------------------------------------- driver
extern "C" void kernel_launch(void* const* d_in, const int* in_sizes, int n_in,
                              void* d_out, int out_size, void* d_ws, size_t ws_size,
                              hipStream_t stream) {
    (void)n_in; (void)out_size; (void)ws_size;

    const float* x   = (const float*)d_in[0];
    const int* edge  = (const int*)d_in[1];
    const float* W1  = (const float*)d_in[2];
    const float* b1  = (const float*)d_in[3];
    const float* W2  = (const float*)d_in[4];
    const float* b2  = (const float*)d_in[5];
    const float* W3  = (const float*)d_in[6];
    const float* b3  = (const float*)d_in[7];
    const float* W4  = (const float*)d_in[8];
    const float* b4  = (const float*)d_in[9];
    const float* Wc  = (const float*)d_in[10];
    const float* bc  = (const float*)d_in[11];
    const float* g1  = (const float*)d_in[12];
    const float* be1 = (const float*)d_in[13];
    const float* g2  = (const float*)d_in[14];
    const float* be2 = (const float*)d_in[15];
    const float* g3  = (const float*)d_in[16];
    const float* be3 = (const float*)d_in[17];
    const float* g4  = (const float*)d_in[18];
    const float* be4 = (const float*)d_in[19];

    const int hid  = in_sizes[3];            // 256
    const int gene = in_sizes[2] / hid;      // 512
    const int N    = in_sizes[0] / gene;     // 50000
    const int hid2 = in_sizes[5];            // 512
    const int hq   = in_sizes[9];            // 128
    const int nc   = in_sizes[11];           // 4
    const int E    = in_sizes[1] / 2;        // 1.6M
    const int* src = edge;
    const int* dst = edge + E;

    // ---- workspace carve-out (256B aligned slabs)
    char* base = (char*)d_ws;
    size_t off = 0;
    auto alloc = [&](size_t bytes) -> char* {
        char* p = base + off;
        off += (bytes + 255) & ~(size_t)255;
        return p;
    };
    float*    deg   = (float*)   alloc((size_t)N * 4);
    float*    dinv  = (float*)   alloc((size_t)N * 4);
    _Float16* wt16  = (_Float16*)alloc((size_t)gene * hid2 * 2);
    _Float16* act16 = (_Float16*)alloc((size_t)N * gene * 2);   // max feature dim
    float*    xw    = (float*)   alloc((size_t)N * hid2 * 4);
    float*    agg   = (float*)   alloc((size_t)N * hid2 * 4);
    float*    h1    = (float*)   alloc((size_t)N * hid * 4);
    float*    h3    = (float*)   alloc((size_t)N * hid * 4);

    float* out_logits = (float*)d_out;                           // [N, nc]
    float* out_conv   = out_logits + (size_t)N * nc;             // [N, hq]
    float* out_bn     = out_conv   + (size_t)N * hq;             // [N, hq]

    const int B = 256;
    const int nBlocksN = (N + B - 1) / B;
    const int nBlocksE = (E + B - 1) / B;

    // ---- normalization coefficients (recomputed every call: deterministic)
    k_init_deg <<<nBlocksN, B, 0, stream>>>(deg, N);
    k_deg_edges<<<nBlocksE, B, 0, stream>>>(dst, deg, E);
    k_dinv     <<<nBlocksN, B, 0, stream>>>(deg, dinv, N);

    // ---- x -> f16
    {
        long n = (long)N * gene;
        int gb = (int)((n + B - 1) / B);
        if (gb > 65535 * 16) gb = 65535 * 16;
        k_f32_to_f16<<<gb, B, 0, stream>>>(x, act16, n);
    }

    auto run_gcn = [&](const _Float16* a16, const float* W, int K, int Nc) {
        k_wT_f16<<<(K * Nc + B - 1) / B, B, 0, stream>>>(W, wt16, K, Nc);
        dim3 grid(Nc / 128, (N + 63) / 64);
        k_gemm_wmma<<<grid, B, 0, stream>>>(a16, wt16, xw, N, K, Nc);
        k_agg_self <<<N, B, 0, stream>>>(dinv, xw, agg, Nc);
        k_agg_edges<<<(E + 7) / 8, B, 0, stream>>>(src, dst, dinv, xw, agg, E, Nc);
    };

    // ---- layer 1: [N,512]x[512,256] -> h1 (f32) + act16 (f16)
    run_gcn(act16, W1, gene, hid);
    k_ln_elu<<<N, B, 0, stream>>>(agg, b1, g1, be1, nullptr, nullptr, h1, act16, hid);

    // ---- layer 2: [N,256]x[256,512] -> act16 (f16 only)
    run_gcn(act16, W2, hid, hid2);
    k_ln_elu<<<N, B, 0, stream>>>(agg, b2, g2, be2, nullptr, nullptr, nullptr, act16, hid2);

    // ---- layer 3: [N,512]x[512,256] -> h3 (f32)
    run_gcn(act16, W3, hid2, hid);
    k_ln_elu<<<N, B, 0, stream>>>(agg, b3, g3, be3, nullptr, nullptr, h3, nullptr, hid);

    // ---- residual: act16 = f16(h3 + h1)
    {
        long n = (long)N * hid;
        k_add_f16<<<(int)((n + B - 1) / B), B, 0, stream>>>(h3, h1, act16, n);
    }

    // ---- layer 4: [N,256]x[256,128] -> out_conv / out_bn straight into d_out
    run_gcn(act16, W4, hid, hq);
    k_ln_elu<<<N, B, 0, stream>>>(agg, b4, g4, be4, out_conv, out_bn,
                                  nullptr, nullptr, hq);

    // ---- classifier: logits = elu(out_bn) @ Wc + bc
    k_classifier<<<nBlocksN, B, 0, stream>>>(out_bn, Wc, bc, out_logits, N, hq, nc);
}
